// STSAGCN_76828374990958
// MI455X (gfx1250) — matrix operand order repaired and checked by reading, not compiled, not run on hardware
//
#include <hip/hip_runtime.h>
#include <math.h>

// ---------------------------------------------------------------------------
// STSAGCN forward for MI455X (gfx1250, wave32, WMMA).
// All contractions use v_wmma_f32_16x16x32_f16 with manually assembled
// fragments per the CDNA5 ISA VGPR layouts (CDNA5 ISA).
// d_in leaf order (setup_inputs dict insertion order, depth-first):
//  0:x 1:p 2..7:emb_o{w1,b1,w3,b3,w2,b2} 8..13:emb_d{...}
//  14..17:gru1{wih,whh,bih,bhh} 18:st.es 19:st.et
//  20..31:sgc{qw,qb,kw,kb,lng,lnb,flg,flb,fw1,fb1,fw2,fb2}
//  32..37:gru2{wih,whh,bih,bhh,ow,ob}
// ---------------------------------------------------------------------------

typedef __attribute__((ext_vector_type(16))) _Float16 v16h;
typedef __attribute__((ext_vector_type(8)))  _Float16 v8h;
typedef __attribute__((ext_vector_type(8)))  float    v8f;

#define DEVI __device__ __forceinline__

DEVI v8f vzero() {
  v8f z;
#pragma unroll
  for (int i = 0; i < 8; ++i) z[i] = 0.f;
  return z;
}

DEVI v8f wmma32(v16h a, v16h b, v8f c) {
  return __builtin_amdgcn_wmma_f32_16x16x32_f16(
      /*neg_a=*/false, a, /*neg_b=*/false, b,
      /*c_mod=*/(short)0, c, /*reuse_a=*/false, /*reuse_b=*/false);
}

// A fragment, 16x32 f16, A row-major with leading dim lda.
// lane L: row = L%16; K-chunks at k0+(L/16)*8 and +16 (ISA 16-bit A layout).
DEVI v16h frag_a(const _Float16* __restrict__ Abase, int lda, int k0, int lane) {
  const _Float16* p = Abase + (size_t)(lane & 15) * lda + k0 + ((lane >> 4) << 3);
  v8h lo = *(const v8h*)p;
  v8h hi = *(const v8h*)(p + 16);
  v16h a;
#pragma unroll
  for (int i = 0; i < 8; ++i) { a[i] = lo[i]; a[i + 8] = hi[i]; }
  return a;
}

// A fragment from transposed source: A[m][k] = X[k][m], X row-major ldx (LDS).
DEVI v16h frag_aT(const _Float16* __restrict__ X, int ldx, int m0, int k0, int lane) {
  int m  = m0 + (lane & 15);
  int ks = k0 + ((lane >> 4) << 3);
  v16h a;
#pragma unroll
  for (int j = 0; j < 8; ++j) {
    a[j]     = X[(ks + j) * ldx + m];
    a[j + 8] = X[(ks + 16 + j) * ldx + m];
  }
  return a;
}

// B fragment, 32x16 f16. colPtr points at this lane's column (column-major,
// i.e. 16+ consecutive K values contiguous). lane L: K-chunks at k0+(L/16)*8 and +16.
DEVI v16h frag_b(const _Float16* __restrict__ colPtr, int k0, int lane) {
  const _Float16* p = colPtr + k0 + ((lane >> 4) << 3);
  v8h lo = *(const v8h*)p;
  v8h hi = *(const v8h*)(p + 16);
  v16h b;
#pragma unroll
  for (int i = 0; i < 8; ++i) { b[i] = lo[i]; b[i + 8] = hi[i]; }
  return b;
}

DEVI float sigm(float x) { return 1.f / (1.f + expf(-x)); }

// ---------------------------------------------------------------------------
// Spatio-temporal embedding gather: pe (B,5,200,64) f16, pe2 (B,12,200,64) f16
// ---------------------------------------------------------------------------
__global__ void st_emb_kernel(const int* __restrict__ p,
                              const float* __restrict__ es,
                              const float* __restrict__ et,
                              _Float16* __restrict__ pe,
                              _Float16* __restrict__ pe2) {
  int t = blockIdx.x * blockDim.x + threadIdx.x;
  const int TOT = 16 * 17 * 200 * 64;
  if (t >= TOT) return;
  int d = t & 63;
  int n = (t >> 6) % 200;
  int s = (t / (64 * 200)) % 17;
  int b = t / (64 * 200 * 17);
  int q;
  if (s < 4)       q = p[b * 60 + 11 + 12 * s];
  else if (s == 4) q = p[b * 60 + 48];
  else             q = p[b * 60 + 48 + (s - 5)];
  float v = es[(size_t)q * 12800 + n * 64 + d] + et[q * 64 + d];
  if (s < 5) pe [(size_t)((b * 5  + s)       * 200 + n) * 64 + d] = (_Float16)v;
  else       pe2[(size_t)((b * 12 + (s - 5)) * 200 + n) * 64 + d] = (_Float16)v;
}

// ---------------------------------------------------------------------------
// Patch-fused embedding: for each (b,patch), both orientations of
// h = (X@w1+b1)*sigmoid(X@w3+b3), reduced over l with w2, +b2.
// e_emb layout (B,5,200,64) f16 : cols 0..31 = e1, 32..63 = e2.
// Dynamic LDS: Xr 224x224 f16 + two 64x224 col-major weight blocks.
// ---------------------------------------------------------------------------
__global__ __launch_bounds__(256) void embed_kernel(
    const float* __restrict__ x,
    const float* __restrict__ w1o, const float* __restrict__ b1o,
    const float* __restrict__ w3o, const float* __restrict__ b3o,
    const float* __restrict__ w2o, const float* __restrict__ b2o,
    const float* __restrict__ w1d, const float* __restrict__ b1d,
    const float* __restrict__ w3d, const float* __restrict__ b3d,
    const float* __restrict__ w2d, const float* __restrict__ b2d,
    _Float16* __restrict__ e_emb) {
  extern __shared__ _Float16 sm[];
  _Float16* Xr = sm;                 // 224*224
  _Float16* Wo = sm + 224 * 224;     // 64 cols * 224
  _Float16* Wd = Wo + 64 * 224;
  const int tid = threadIdx.x, lane = tid & 31, wave = tid >> 5;
  const int b = blockIdx.x / 5, pp = blockIdx.x % 5;

  for (int i = tid; i < 224 * 224; i += 256) Xr[i] = (_Float16)0.f;
  for (int i = tid; i < 64 * 224; i += 256) {
    int c = i / 224, k = i % 224;
    float vo = 0.f, vd = 0.f;
    if (k < 200) {
      vo = (c < 32) ? w1o[k * 32 + c] : w3o[k * 32 + (c - 32)];
      vd = (c < 32) ? w1d[k * 32 + c] : w3d[k * 32 + (c - 32)];
    }
    Wo[c * 224 + k] = (_Float16)vo;
    Wd[c * 224 + k] = (_Float16)vd;
  }
  __syncthreads();

  // jobs: (orientation, mtile) pairs; 26 total, strided over 8 waves.
  int jobs[4], njob = 0;
  for (int j = wave; j < 26; j += 8) jobs[njob++] = j;

  float acc[4][2][8];
  float b1v[4][2], b3v[4][2];
  for (int ji = 0; ji < njob; ++ji) {
    int orient = jobs[ji] & 1;
    for (int nt = 0; nt < 2; ++nt) {
      int col = nt * 16 + (lane & 15);
      b1v[ji][nt] = orient ? b1d[col] : b1o[col];
      b3v[ji][nt] = orient ? b3d[col] : b3o[col];
      for (int i = 0; i < 8; ++i) acc[ji][nt][i] = 0.f;
    }
  }

  for (int l = 0; l < 12; ++l) {
    int tt = pp * 12 + l; if (tt > 47) tt = 47;
    const float* src = x + (size_t)(b * 48 + tt) * 40000;
    for (int q = tid; q < 10000; q += 256) {
      int r = q / 50, c4 = (q % 50) * 4;
      __builtin_prefetch(src + r * 200 + c4 + 6400, 0, 1);
      const float4 v = *(const float4*)(src + r * 200 + c4);
      _Float16* d = Xr + r * 224 + c4;
      d[0] = (_Float16)v.x; d[1] = (_Float16)v.y;
      d[2] = (_Float16)v.z; d[3] = (_Float16)v.w;
    }
    __syncthreads();
    float wlo = w2o[l], wld = w2d[l];
    for (int ji = 0; ji < njob; ++ji) {
      int job = jobs[ji];
      int orient = job & 1, m0 = (job >> 1) * 16;
      v8f c[4];
      for (int nt = 0; nt < 4; ++nt) c[nt] = vzero();
      const _Float16* W = orient ? Wd : Wo;
      for (int k0 = 0; k0 < 224; k0 += 32) {
        v16h a = orient ? frag_aT(Xr, 224, m0, k0, lane)
                        : frag_a(Xr + m0 * 224, 224, k0, lane);
#pragma unroll
        for (int nt = 0; nt < 4; ++nt) {
          v16h bb = frag_b(W + (nt * 16 + (lane & 15)) * 224, k0, lane);
          c[nt] = wmma32(a, bb, c[nt]);
        }
      }
      float wl = orient ? wld : wlo;
      for (int nt = 0; nt < 2; ++nt)
        for (int i = 0; i < 8; ++i) {
          float h1 = c[nt][i] + b1v[ji][nt];
          float h3 = c[nt + 2][i] + b3v[ji][nt];
          acc[ji][nt][i] += wl * h1 * sigm(h3);
        }
    }
    __syncthreads();
  }
  for (int ji = 0; ji < njob; ++ji) {
    int job = jobs[ji];
    int orient = job & 1, m0 = (job >> 1) * 16;
    float b2 = orient ? b2d[0] : b2o[0];
    for (int nt = 0; nt < 2; ++nt) {
      int col = nt * 16 + (lane & 15) + (orient ? 32 : 0);
      for (int i = 0; i < 8; ++i) {
        int row = m0 + i + 8 * (lane >> 4);
        if (row < 200)
          e_emb[(size_t)((b * 5 + pp) * 200 + row) * 64 + col] =
              (_Float16)(acc[ji][nt][i] + b2);
      }
    }
  }
}

// ---------------------------------------------------------------------------
// One GRU step: Hout = gate(Xin@wih+bih, Hprev@whh+bhh, Hprev).
// Per-batch rows = 200 (B=16 batches), all matmuls WMMA, K=64.
// ---------------------------------------------------------------------------
__global__ __launch_bounds__(256) void gru_step_kernel(
    const _Float16* __restrict__ Xin, int xBS,
    const _Float16* __restrict__ Hprev, int hBS,
    _Float16* __restrict__ Hout, int hoBS,
    const float* __restrict__ wih, const float* __restrict__ whh,
    const float* __restrict__ bih, const float* __restrict__ bhh) {
  __shared__ _Float16 Wi[192 * 64];
  __shared__ _Float16 Wh[192 * 64];
  int tid = threadIdx.x, lane = tid & 31, wave = tid >> 5;
  for (int i = tid; i < 192 * 64; i += 256) {
    int c = i >> 6, k = i & 63;
    Wi[c * 64 + k] = (_Float16)wih[k * 192 + c];
    Wh[c * 64 + k] = (_Float16)whh[k * 192 + c];
  }
  __syncthreads();
  int gid = blockIdx.x * 8 + wave, nw = gridDim.x * 8;
  for (int job = gid; job < 16 * 13 * 4; job += nw) {
    int b = job / 52, r = job % 52, mtile = r >> 2, g = r & 3;
    int m0 = mtile * 16;
    v8f air = vzero(), aiz = vzero(), aic = vzero();
    v8f ahr = vzero(), ahz = vzero(), ahc = vzero();
    const _Float16* Xb = Xin   + (size_t)b * xBS + (size_t)m0 * 64;
    const _Float16* Hb = Hprev + (size_t)b * hBS + (size_t)m0 * 64;
    int cb = g * 16 + (lane & 15);
    for (int k0 = 0; k0 < 64; k0 += 32) {
      v16h ax = frag_a(Xb, 64, k0, lane);
      v16h ah = frag_a(Hb, 64, k0, lane);
      air = wmma32(ax, frag_b(Wi + (cb      ) * 64, k0, lane), air);
      aiz = wmma32(ax, frag_b(Wi + (cb +  64) * 64, k0, lane), aiz);
      aic = wmma32(ax, frag_b(Wi + (cb + 128) * 64, k0, lane), aic);
      ahr = wmma32(ah, frag_b(Wh + (cb      ) * 64, k0, lane), ahr);
      ahz = wmma32(ah, frag_b(Wh + (cb +  64) * 64, k0, lane), ahz);
      ahc = wmma32(ah, frag_b(Wh + (cb + 128) * 64, k0, lane), ahc);
    }
    int j = g * 16 + (lane & 15);
    float bir = bih[j], biz = bih[j + 64], bic = bih[j + 128];
    float bhr = bhh[j], bhz = bhh[j + 64], bhc = bhh[j + 128];
    for (int i = 0; i < 8; ++i) {
      int row = m0 + i + 8 * (lane >> 4);
      if (row < 200) {
        float rr = sigm(air[i] + bir + ahr[i] + bhr);
        float zz = sigm(aiz[i] + biz + ahz[i] + bhz);
        float cc = tanhf(aic[i] + bic + rr * (ahc[i] + bhc));
        float hv = (float)Hprev[(size_t)b * hBS + row * 64 + j];
        Hout[(size_t)b * hoBS + row * 64 + j] = (_Float16)((1.f - zz) * cc + zz * hv);
      }
    }
  }
}

// ---------------------------------------------------------------------------
// LayerNorm over 64 features, one wave per row (wave32: 2 elems/lane).
// ---------------------------------------------------------------------------
__global__ void ln_kernel(const _Float16* __restrict__ in, _Float16* __restrict__ out,
                          const float* __restrict__ g, const float* __restrict__ bt,
                          int rows) {
  int lane = threadIdx.x & 31, wave = threadIdx.x >> 5;
  int row = blockIdx.x * 8 + wave;
  if (row >= rows) return;
  float x0 = (float)in[(size_t)row * 64 + lane];
  float x1 = (float)in[(size_t)row * 64 + lane + 32];
  float s = x0 + x1, q = x0 * x0 + x1 * x1;
  for (int m = 16; m >= 1; m >>= 1) { s += __shfl_xor(s, m); q += __shfl_xor(q, m); }
  float mean = s * (1.f / 64.f);
  float var  = q * (1.f / 64.f) - mean * mean;
  float rstd = rsqrtf(var + 1e-5f);
  out[(size_t)row * 64 + lane]      = (_Float16)((x0 - mean) * rstd * g[lane]      + bt[lane]);
  out[(size_t)row * 64 + lane + 32] = (_Float16)((x1 - mean) * rstd * g[lane + 32] + bt[lane + 32]);
}

// ---------------------------------------------------------------------------
// Generic WMMA GEMM: C = A(f16,R x K) @ B(f32 global, K x N) + bias
//   (+ optional residual, optional relu); out f16 and/or f32.
// B is transposed into LDS (column-major, zero-padded to Npad).
// ---------------------------------------------------------------------------
__global__ __launch_bounds__(256) void gemm_kernel(
    const _Float16* __restrict__ A, int aBS, int rowsPB, int nbatch,
    const float* __restrict__ Bg, int K, int Ncols,
    const float* __restrict__ bias,
    const float* __restrict__ resid, int resBS, int relu,
    _Float16* __restrict__ Cf16, float* __restrict__ Cf32, int cBS, int ldc) {
  extern __shared__ _Float16 Bl[];
  int tid = threadIdx.x, lane = tid & 31, wave = tid >> 5;
  int ntc = (Ncols + 15) >> 4, Npad = ntc << 4;
  for (int i = tid; i < Npad * K; i += 256) Bl[i] = (_Float16)0.f;
  __syncthreads();
  for (int i = tid; i < K * Ncols; i += 256) {
    int k = i / Ncols, c = i % Ncols;
    Bl[c * K + k] = (_Float16)Bg[i];
  }
  __syncthreads();
  int mtc = (rowsPB + 15) >> 4;
  int total = nbatch * mtc * ntc;
  int gid = blockIdx.x * 8 + wave, nw = gridDim.x * 8;
  for (int job = gid; job < total; job += nw) {
    int b = job / (mtc * ntc), r = job % (mtc * ntc), mt = r / ntc, nt = r % ntc;
    int m0 = mt * 16;
    v8f acc = vzero();
    const _Float16* Ab = A + (size_t)b * aBS + (size_t)m0 * K;
    const _Float16* Bc = Bl + (nt * 16 + (lane & 15)) * K;
    for (int k0 = 0; k0 < K; k0 += 32)
      acc = wmma32(frag_a(Ab, K, k0, lane), frag_b(Bc, k0, lane), acc);
    int col = nt * 16 + (lane & 15);
    if (col < Ncols) {
      float bv = bias ? bias[col] : 0.f;
      for (int i = 0; i < 8; ++i) {
        int row = m0 + i + 8 * (lane >> 4);
        if (row < rowsPB) {
          float v = acc[i] + bv;
          if (resid) v += resid[(size_t)b * resBS + (size_t)row * ldc + col];
          if (relu)  v = fmaxf(v, 0.f);
          size_t idx = (size_t)b * cBS + (size_t)row * ldc + col;
          if (Cf16) Cf16[idx] = (_Float16)v;
          if (Cf32) Cf32[idx] = v;
        }
      }
    }
  }
}

// ---------------------------------------------------------------------------
// Attention scores: sag = sigmoid(0.125*Q.K^T) * (P4.PE^T) * 0.125
// Q,P4: (B,200,64) f16; K_,PE: (B,1000,64) f16 (B^T columns are contiguous rows)
// ---------------------------------------------------------------------------
__global__ __launch_bounds__(256) void scores_kernel(
    const _Float16* __restrict__ q_, const _Float16* __restrict__ k_,
    const _Float16* __restrict__ pe, float* __restrict__ scores) {
  int lane = threadIdx.x & 31, wave = threadIdx.x >> 5;
  int gid = blockIdx.x * 8 + wave, nw = gridDim.x * 8;
  const int total = 16 * 13 * 63;
  for (int job = gid; job < total; job += nw) {
    int b = job / (13 * 63), r = job % (13 * 63), mt = r / 63, nt = r % 63;
    int m0 = mt * 16, n0 = nt * 16;
    v8f aD = vzero(), aS = vzero();
    const _Float16* Qb = q_ + ((size_t)b * 200 + m0) * 64;
    const _Float16* Pq = pe + ((size_t)(b * 5 + 4) * 200 + m0) * 64;
    int tm = n0 + (lane & 15);
    const _Float16* Kc = k_ + ((size_t)b * 1000 + tm) * 64;
    const _Float16* Pc = pe + ((size_t)b * 1000 + tm) * 64;
    for (int k0 = 0; k0 < 64; k0 += 32) {
      aD = wmma32(frag_a(Qb, 64, k0, lane), frag_b(Kc, k0, lane), aD);
      aS = wmma32(frag_a(Pq, 64, k0, lane), frag_b(Pc, k0, lane), aS);
    }
    if (tm < 1000) {
      for (int i = 0; i < 8; ++i) {
        int row = m0 + i + 8 * (lane >> 4);
        if (row < 200)
          scores[((size_t)b * 200 + row) * 1000 + tm] =
              sigm(aD[i] * 0.125f) * aS[i] * 0.125f;
      }
    }
  }
}

// ---------------------------------------------------------------------------
// Exact top-600 threshold (bitwise radix select on monotone float keys) +
// masked softmax. One wave32 per row of 1000; probs stored f16 stride 1024.
// ---------------------------------------------------------------------------
__global__ __launch_bounds__(256) void topk_softmax_kernel(
    const float* __restrict__ scores, _Float16* __restrict__ sag_p) {
  int lane = threadIdx.x & 31, wave = threadIdx.x >> 5;
  int row = blockIdx.x * 8 + wave;
  if (row >= 3200) return;
  const float* base = scores + (size_t)row * 1000;
  float v[32]; unsigned key[32];
  for (int i = 0; i < 32; ++i) {
    int idx = lane + (i << 5);
    float f = (idx < 1000) ? base[idx] : -3.0e38f;
    v[i] = f;
    unsigned bits = __float_as_uint(f);
    key[i] = (bits & 0x80000000u) ? ~bits : (bits | 0x80000000u);
  }
  unsigned T = 0;
  for (int bit = 31; bit >= 0; --bit) {
    unsigned cand = T | (1u << bit);
    int c = 0;
    for (int i = 0; i < 32; ++i) c += (key[i] >= cand) ? 1 : 0;
    for (int m = 16; m >= 1; m >>= 1) c += __shfl_xor(c, m);
    if (c >= 600) T = cand;   // T = 600th-largest key
  }
  float mx = -3.0e38f;
  for (int i = 0; i < 32; ++i) mx = fmaxf(mx, v[i]);
  for (int m = 16; m >= 1; m >>= 1) mx = fmaxf(mx, __shfl_xor(mx, m));
  float sum = 0.f;
  for (int i = 0; i < 32; ++i) {
    float e = (key[i] > T) ? __expf(v[i] - mx) : 0.f;  // strict >, as reference
    v[i] = e; sum += e;
  }
  for (int m = 16; m >= 1; m >>= 1) sum += __shfl_xor(sum, m);
  float inv = (sum > 0.f) ? 1.f / sum : 0.f;
  for (int i = 0; i < 32; ++i) {
    int idx = lane + (i << 5);
    sag_p[(size_t)row * 1024 + idx] = (_Float16)(v[i] * inv);
  }
}

// ---------------------------------------------------------------------------
// o = sag @ x_, z = o + y_. One block per b; x_^T staged in LDS (64x1024 f16).
// ---------------------------------------------------------------------------
__global__ __launch_bounds__(256) void ogemm_kernel(
    const _Float16* __restrict__ sag_p, const _Float16* __restrict__ xln,
    float* __restrict__ zf32, _Float16* __restrict__ zf16) {
  extern __shared__ _Float16 Xt[];   // 64 * 1024
  int tid = threadIdx.x, lane = tid & 31, wave = tid >> 5;
  int b = blockIdx.x;
  for (int i = tid; i < 64 * 1024; i += 256) Xt[i] = (_Float16)0.f;
  __syncthreads();
  for (int i = tid; i < 1000 * 64; i += 256) {
    int rr = i >> 6, d = i & 63;
    Xt[d * 1024 + rr] = xln[((size_t)b * 1000 + rr) * 64 + d];
  }
  __syncthreads();
  for (int mt = wave; mt < 13; mt += 8) {
    int m0 = mt * 16;
    v8f acc[4];
    for (int nt = 0; nt < 4; ++nt) acc[nt] = vzero();
    const _Float16* Ab = sag_p + ((size_t)b * 200 + m0) * 1024;
    for (int k0 = 0; k0 < 1024; k0 += 32) {
      v16h a = frag_a(Ab, 1024, k0, lane);
#pragma unroll
      for (int nt = 0; nt < 4; ++nt)
        acc[nt] = wmma32(a, frag_b(Xt + (nt * 16 + (lane & 15)) * 1024, k0, lane), acc[nt]);
    }
    for (int nt = 0; nt < 4; ++nt) {
      int col = nt * 16 + (lane & 15);
      for (int i = 0; i < 8; ++i) {
        int row = m0 + i + 8 * (lane >> 4);
        if (row < 200) {
          float y = (float)xln[((size_t)(b * 5 + 4) * 200 + row) * 64 + col];
          float v = acc[nt][i] + y;
          size_t idx = ((size_t)b * 200 + row) * 64 + col;
          zf32[idx] = v; zf16[idx] = (_Float16)v;
        }
      }
    }
  }
}

// ---------------------------------------------------------------------------
extern "C" void kernel_launch(void* const* d_in, const int* in_sizes, int n_in,
                              void* d_out, int out_size, void* d_ws, size_t ws_size,
                              hipStream_t stream) {
  (void)in_sizes; (void)n_in; (void)out_size; (void)ws_size;
  const float* x    = (const float*)d_in[0];
  const int*   p    = (const int*)  d_in[1];
  const float* w1o  = (const float*)d_in[2];
  const float* b1o  = (const float*)d_in[3];
  const float* w3o  = (const float*)d_in[4];
  const float* b3o  = (const float*)d_in[5];
  const float* w2o  = (const float*)d_in[6];
  const float* b2o  = (const float*)d_in[7];
  const float* w1d  = (const float*)d_in[8];
  const float* b1d  = (const float*)d_in[9];
  const float* w3d  = (const float*)d_in[10];
  const float* b3d  = (const float*)d_in[11];
  const float* w2d  = (const float*)d_in[12];
  const float* b2d  = (const float*)d_in[13];
  const float* g1wih= (const float*)d_in[14];
  const float* g1whh= (const float*)d_in[15];
  const float* g1bih= (const float*)d_in[16];
  const float* g1bhh= (const float*)d_in[17];
  const float* es   = (const float*)d_in[18];
  const float* et   = (const float*)d_in[19];
  const float* qw   = (const float*)d_in[20];
  const float* qb   = (const float*)d_in[21];
  const float* kw   = (const float*)d_in[22];
  const float* kb   = (const float*)d_in[23];
  const float* lng  = (const float*)d_in[24];
  const float* lnb  = (const float*)d_in[25];
  const float* flg  = (const float*)d_in[26];
  const float* flb  = (const float*)d_in[27];
  const float* fw1  = (const float*)d_in[28];
  const float* fb1  = (const float*)d_in[29];
  const float* fw2  = (const float*)d_in[30];
  const float* fb2  = (const float*)d_in[31];
  const float* g2wih= (const float*)d_in[32];
  const float* g2whh= (const float*)d_in[33];
  const float* g2bih= (const float*)d_in[34];
  const float* g2bhh= (const float*)d_in[35];
  const float* ow   = (const float*)d_in[36];
  const float* ob   = (const float*)d_in[37];

  // workspace carve (each block gets 8KB slack for masked-tile overreads)
  char* w = (char*)d_ws; size_t off = 0;
  auto carve = [&](size_t bytes) -> void* {
    void* r = w + off;
    off = (off + bytes + 8192 + 1023) & ~(size_t)1023;
    return r;
  };
  _Float16* e_emb = (_Float16*)carve((size_t)1024000 * 2);   // (B,5,200,64)
  _Float16* h_seq = (_Float16*)carve((size_t)1024000 * 2);   // gru1 states
  _Float16* xln   = (_Float16*)carve((size_t)1024000 * 2);
  _Float16* q_    = (_Float16*)carve((size_t)204800  * 2);
  _Float16* k_    = (_Float16*)carve((size_t)1024000 * 2);
  _Float16* pe    = (_Float16*)carve((size_t)1024000 * 2);
  _Float16* pe2   = (_Float16*)carve((size_t)2457600 * 2);
  float*    sc    = (float*)   carve((size_t)3200000 * 4);
  _Float16* sagp  = (_Float16*)carve((size_t)3276800 * 2);
  float*    zf32  = (float*)   carve((size_t)204800  * 4);
  _Float16* zf16  = (_Float16*)carve((size_t)204800  * 2);
  _Float16* lnz   = (_Float16*)carve((size_t)204800  * 2);
  _Float16* t2    = (_Float16*)carve((size_t)819200  * 2);
  _Float16* enc   = (_Float16*)carve((size_t)204800  * 2);
  _Float16* hsq2  = (_Float16*)carve((size_t)2457600 * 2);
  _Float16* zbuf  = (_Float16*)carve((size_t)204800  * 2);

  hipMemsetAsync(zbuf, 0, (size_t)204800 * 2, stream);

  hipFuncSetAttribute(reinterpret_cast<const void*>(&embed_kernel),
                      hipFuncAttributeMaxDynamicSharedMemorySize, 180 * 1024);
  hipFuncSetAttribute(reinterpret_cast<const void*>(&ogemm_kernel),
                      hipFuncAttributeMaxDynamicSharedMemorySize, 160 * 1024);

  { int tot = 16 * 17 * 200 * 64;
    st_emb_kernel<<<(tot + 255) / 256, 256, 0, stream>>>(p, es, et, pe, pe2); }

  size_t embLds = (size_t)(224 * 224 + 2 * 64 * 224) * sizeof(_Float16);
  embed_kernel<<<80, 256, embLds, stream>>>(x, w1o, b1o, w3o, b3o, w2o, b2o,
                                            w1d, b1d, w3d, b3d, w2d, b2d, e_emb);

  for (int t = 0; t < 5; ++t) {
    const _Float16* hp = (t == 0) ? zbuf : (h_seq + (size_t)(t - 1) * 200 * 64);
    int hpBS = (t == 0) ? 200 * 64 : 5 * 200 * 64;
    gru_step_kernel<<<104, 256, 0, stream>>>(
        e_emb + (size_t)t * 200 * 64, 5 * 200 * 64, hp, hpBS,
        h_seq + (size_t)t * 200 * 64, 5 * 200 * 64, g1wih, g1whh, g1bih, g1bhh);
  }

  ln_kernel<<<2000, 256, 0, stream>>>(h_seq, xln, lng, lnb, 16000);

  gemm_kernel<<<128, 256, 64 * 64 * 2, stream>>>(          // q_ = x_[:, -1]@qw+qb
      xln + 4 * 200 * 64, 5 * 200 * 64, 200, 16,
      qw, 64, 64, qb, nullptr, 0, 0, q_, nullptr, 200 * 64, 64);
  gemm_kernel<<<512, 256, 64 * 64 * 2, stream>>>(          // k_ = x_@kw+kb
      xln, 0, 16000, 1, kw, 64, 64, kb, nullptr, 0, 0, k_, nullptr, 0, 64);

  scores_kernel<<<512, 256, 0, stream>>>(q_, k_, pe, sc);
  topk_softmax_kernel<<<400, 256, 0, stream>>>(sc, sagp);
  ogemm_kernel<<<16, 256, 64 * 1024 * 2, stream>>>(sagp, xln, zf32, zf16);
  ln_kernel<<<400, 256, 0, stream>>>(zf16, lnz, flg, flb, 3200);
  gemm_kernel<<<400, 256, 256 * 64 * 2, stream>>>(         // t2 = relu(lnz@fw1+fb1)
      lnz, 0, 3200, 1, fw1, 64, 256, fb1, nullptr, 0, 1, t2, nullptr, 0, 256);
  gemm_kernel<<<400, 256, 64 * 256 * 2, stream>>>(         // enc = z + t2@fw2+fb2
      t2, 0, 3200, 1, fw2, 256, 64, fb2, zf32, 0, 0, enc, nullptr, 0, 64);

  for (int t = 0; t < 12; ++t) {
    const _Float16* hp = (t == 0) ? enc : (hsq2 + (size_t)(t - 1) * 200 * 64);
    int hpBS = (t == 0) ? 200 * 64 : 12 * 200 * 64;
    gru_step_kernel<<<104, 256, 0, stream>>>(
        pe2 + (size_t)t * 200 * 64, 12 * 200 * 64, hp, hpBS,
        hsq2 + (size_t)t * 200 * 64, 12 * 200 * 64, g2wih, g2whh, g2bih, g2bhh);
  }

  gemm_kernel<<<1024, 256, 208 * 64 * 2, stream>>>(        // out = hs@ow+ob (f32)
      hsq2, 0, 38400, 1, ow, 64, 200, ob, nullptr, 0, 0,
      nullptr, (float*)d_out, 0, 200);
}
